// CURAttention_72103910965568
// MI455X (gfx1250) — compile-verified, matrix operand order checked by reading, not run on hardware
//
#include <hip/hip_runtime.h>
#include <hip/hip_bf16.h>
#include <cfloat>
#include <cstdint>

// Problem constants (from reference)
#define B_   2
#define H_   8
#define N_   4096
#define D_   64
#define MSEL 256
#define BH   (B_ * H_)

typedef __attribute__((ext_vector_type(16))) __bf16 v16bf;
typedef __attribute__((ext_vector_type(8)))  float  v8f;
typedef __attribute__((ext_vector_type(2)))  float  v2f;
typedef __attribute__((ext_vector_type(4)))  int    v4i;

union FragBF { v16bf v; uint4 q[2]; };

#if defined(__AMDGCN__) && __has_builtin(__builtin_amdgcn_global_load_async_to_lds_b128)
#define HAS_ASYNC_LDS 1
typedef __attribute__((address_space(1))) v4i* gv4i_p;
typedef __attribute__((address_space(3))) v4i* lv4i_p;
#else
#define HAS_ASYNC_LDS 0
#endif

// ---------------------------------------------------------------------------
// 1. Convert: Qs = Q/8 -> bf16, K -> bf16, V -> V^T bf16 ([BH,64,4096])
// ---------------------------------------------------------------------------
__global__ void convert_kernel(const float* __restrict__ Q, const float* __restrict__ K,
                               const float* __restrict__ V, __bf16* __restrict__ Qs,
                               __bf16* __restrict__ Kb, __bf16* __restrict__ VT) {
  long i = (long)blockIdx.x * blockDim.x + threadIdx.x;
  const long total = (long)BH * N_ * D_;
  if (i >= total) return;
  long bh  = i / (N_ * D_);
  long rem = i - bh * (N_ * D_);
  long n   = rem / D_;
  long d   = rem - n * D_;
  Qs[i] = (__bf16)(Q[i] * 0.125f);
  Kb[i] = (__bf16)K[i];
  VT[bh * (long)(D_ * N_) + d * N_ + n] = (__bf16)V[i];
}

// ---------------------------------------------------------------------------
// 2. Masked row sums (for top-k selection)
// ---------------------------------------------------------------------------
__global__ void rowsum_kernel(const float* __restrict__ T, const unsigned char* __restrict__ mask,
                              float* __restrict__ out, float scale) {
  int r = blockIdx.x * blockDim.x + threadIdx.x;  // [0, BH*N)
  if (r >= BH * N_) return;
  int bh = r / N_;
  int n  = r - bh * N_;
  int b  = bh / H_;
  const float* row = T + (long)r * D_;
  float s = 0.f;
  #pragma unroll
  for (int d = 0; d < D_; ++d) s += row[d];
  out[r] = mask[b * N_ + n] ? s * scale : -FLT_MAX;
}

// ---------------------------------------------------------------------------
// 3. Top-256 per (b,h): iterative argmax, smaller index wins ties
// ---------------------------------------------------------------------------
__global__ void topk_kernel(const float* __restrict__ sums, int* __restrict__ idx) {
  __shared__ float sv[N_];
  __shared__ float rv[256];
  __shared__ int   ri[256];
  int tid = threadIdx.x;
  int bh  = blockIdx.x;
  for (int i = tid; i < N_; i += 256) sv[i] = sums[bh * N_ + i];
  __syncthreads();
  for (int sel = 0; sel < MSEL; ++sel) {
    float bv = -FLT_MAX; int bi = 0x7fffffff;
    for (int i = tid; i < N_; i += 256) {
      float v = sv[i];
      if (v > bv || (v == bv && i < bi)) { bv = v; bi = i; }
    }
    rv[tid] = bv; ri[tid] = bi;
    __syncthreads();
    for (int s = 128; s > 0; s >>= 1) {
      if (tid < s) {
        float ov = rv[tid + s]; int oi = ri[tid + s];
        if (ov > rv[tid] || (ov == rv[tid] && oi < ri[tid])) { rv[tid] = ov; ri[tid] = oi; }
      }
      __syncthreads();
    }
    if (tid == 0) { idx[bh * MSEL + sel] = ri[0]; sv[ri[0]] = -FLT_MAX; }
    __syncthreads();
  }
}

// ---------------------------------------------------------------------------
// 4. Row gather (bf16 rows of length 64)
// ---------------------------------------------------------------------------
__global__ void gather_kernel(const __bf16* __restrict__ src, const int* __restrict__ idx,
                              __bf16* __restrict__ dst) {
  int blk = blockIdx.x;          // [0, BH*MSEL)
  int bh  = blk / MSEL;
  int row = idx[blk];
  dst[(long)blk * D_ + threadIdx.x] = src[((long)bh * N_ + row) * D_ + threadIdx.x];
}

// ---------------------------------------------------------------------------
// 5. Generic bf16 WMMA GEMM:  C[M,N] (f32) = A[M,K] @ Brows[N,K]^T, batched (z)
//    block = 128 threads (4 waves); all waves share one 16-row A tile.
//    A is staged per 64-K chunk (2KB) into LDS via async global->LDS DMA
//    (ASYNCcnt) — each of the 128 threads issues one b128 DMA — double
//    buffered with one barrier per chunk; 2 WMMA k-steps per barrier.
//    B streams straight from global as b128 pairs into WMMA.
// ---------------------------------------------------------------------------
__global__ void gemm_bf16_kernel(const __bf16* __restrict__ A, const __bf16* __restrict__ Brows,
                                 float* __restrict__ C, int M, int N, int K,
                                 long sA, long sB, long sC) {
  (void)M;
  __shared__ __bf16 As[2][16 * 64];   // 2 x 2KB double buffer
  int bh   = blockIdx.z;
  int tid  = threadIdx.x;
  int lane = tid & 31;
  int wave = tid >> 5;
  int m0   = blockIdx.y * 16;
  int n0   = blockIdx.x * 64 + wave * 16;
  int half = lane >> 4;
  int l15  = lane & 15;

  const __bf16* gA = A + bh * sA + (long)m0 * K;                          // 16 x K tile
  const __bf16* pb = Brows + bh * sB + (long)(n0 + l15) * K + half * 8;

  // staging assignment: row = tid/8 (0..15), 16B chunk = tid%8 (0..7)
  int srow = tid >> 3;
  int schk = (tid & 7) * 8;  // in bf16 elements

  v8f acc = {};
  int p = 0;
  for (int ks = 0; ks < K; ks += 64, p ^= 1) {
    const __bf16* gsrc = gA + (long)srow * K + ks + schk;
#if HAS_ASYNC_LDS
    __builtin_amdgcn_global_load_async_to_lds_b128(
        (gv4i_p)(v4i*)gsrc,
        (lv4i_p)(v4i*)&As[p][srow * 64 + schk],
        0, 0);
    asm volatile("s_wait_asynccnt 0x0" ::: "memory");
#else
    *(uint4*)&As[p][srow * 64 + schk] = *(const uint4*)gsrc;
#endif
    __syncthreads();
    #pragma unroll
    for (int kk = 0; kk < 64; kk += 32) {
      FragBF a, b;
      a.q[0] = *(const uint4*)&As[p][l15 * 64 + kk + half * 8];
      a.q[1] = *(const uint4*)&As[p][l15 * 64 + kk + half * 8 + 16];
      b.q[0] = *(const uint4*)(pb);
      b.q[1] = *(const uint4*)(pb + 16);
      pb += 32;
      acc = __builtin_amdgcn_wmma_f32_16x16x32_bf16(false, a.v, false, b.v,
                                                    (short)0, acc, false, false);
    }
  }
  float* pc = C + bh * sC + (long)(m0 + half * 8) * N + n0 + l15;
  #pragma unroll
  for (int j = 0; j < 8; ++j) pc[(long)j * N] = acc[j];
}

// ---------------------------------------------------------------------------
// 6. Row softmax (optionally column-masked), f32 in -> OUT (bf16 or f32)
//    cols <= 4096 so each of 256 threads caches its <=16 values in registers
// ---------------------------------------------------------------------------
template <typename OUT>
__global__ void softmax_kernel(const float* __restrict__ in, OUT* __restrict__ out, int cols,
                               const unsigned char* __restrict__ mask, int rowsPerBH) {
  __shared__ float red[256];
  float lv[16];
  int tid  = threadIdx.x;
  long row = blockIdx.x;
  int bh   = (int)(row / rowsPerBH);
  int b    = bh / H_;
  const float* pin = in + row * cols;
  int nv = cols >> 8;  // cols / 256
  float mx = -FLT_MAX;
  for (int t = 0; t < nv; ++t) {
    int i = tid + (t << 8);
    float v = pin[i];
    if (mask && !mask[b * N_ + i]) v = -FLT_MAX;
    lv[t] = v;
    mx = fmaxf(mx, v);
  }
  red[tid] = mx; __syncthreads();
  for (int s = 128; s > 0; s >>= 1) { if (tid < s) red[tid] = fmaxf(red[tid], red[tid + s]); __syncthreads(); }
  mx = red[0]; __syncthreads();
  float sum = 0.f;
  for (int t = 0; t < nv; ++t) { lv[t] = __expf(lv[t] - mx); sum += lv[t]; }
  red[tid] = sum; __syncthreads();
  for (int s = 128; s > 0; s >>= 1) { if (tid < s) red[tid] += red[tid + s]; __syncthreads(); }
  float inv = 1.f / red[0];
  OUT* pout = out + row * cols;
  for (int t = 0; t < nv; ++t) pout[tid + (t << 8)] = (OUT)(lv[t] * inv);
}

// ---------------------------------------------------------------------------
// 7. Newton-Schulz init: V0 = u^T / max_j(sum_i u_ij)
// ---------------------------------------------------------------------------
__global__ void init_v0_kernel(const float* __restrict__ u, float* __restrict__ V) {
  __shared__ float red[256];
  int j = threadIdx.x;
  const float* U = u + (long)blockIdx.x * MSEL * MSEL;
  float s = 0.f;
  for (int i = 0; i < MSEL; ++i) s += U[(long)i * MSEL + j];
  red[j] = s; __syncthreads();
  for (int st = 128; st > 0; st >>= 1) { if (j < st) red[j] = fmaxf(red[j], red[j + st]); __syncthreads(); }
  float scale = 1.f / red[0];
  float* Vo = V + (long)blockIdx.x * MSEL * MSEL;
  for (int i = 0; i < MSEL; ++i) Vo[(long)j * MSEL + i] = scale * U[(long)i * MSEL + j];
}

// ---------------------------------------------------------------------------
// 8. f32 WMMA GEMM:  C[M,N] = alpha * A[M,K] @ B[K,N] (direct B), batched (z)
// ---------------------------------------------------------------------------
__global__ void gemm_f32_kernel(const float* __restrict__ A, const float* __restrict__ B,
                                float* __restrict__ C, int M, int N, int K, float alpha,
                                long sA, long sB, long sC) {
  (void)M;
  int bh   = blockIdx.z;
  int lane = threadIdx.x & 31;
  int wave = threadIdx.x >> 5;
  int m0   = blockIdx.y * 16;
  int n0   = blockIdx.x * 64 + wave * 16;
  int half = lane >> 4;
  int l15  = lane & 15;
  const float* pa = A + bh * sA + (long)(m0 + l15) * K + half * 2;
  const float* pb = B + bh * sB + (long)(half * 2) * N + n0 + l15;
  v8f acc = {};
  for (int ks = 0; ks < K; ks += 4) {
    __builtin_prefetch((const void*)(pb + (long)16 * N), 0, 0);  // pull next rows into GL2
    v2f a = *(const v2f*)pa;
    v2f b;
    b[0] = pb[0];
    b[1] = pb[N];
    pa += 4;
    pb += (long)4 * N;
    acc = __builtin_amdgcn_wmma_f32_16x16x4_f32(false, a, false, b,
                                                (short)0, acc, false, false);
  }
  float* pc = C + bh * sC + (long)(m0 + half * 8) * N + n0 + l15;
  #pragma unroll
  for (int j = 0; j < 8; ++j) pc[(long)j * N] = alpha * acc[j];
}

// ---------------------------------------------------------------------------
// 9. out = alpha*I - in  (256x256 matrices, batched flat)
// ---------------------------------------------------------------------------
__global__ void ident_minus_kernel(const float* __restrict__ in, float* __restrict__ out, float alpha) {
  long i = (long)blockIdx.x * blockDim.x + threadIdx.x;
  const long total = (long)BH * MSEL * MSEL;
  if (i >= total) return;
  int p = (int)(i % (MSEL * MSEL));
  int r = p >> 8, c = p & 255;
  out[i] = (r == c ? alpha : 0.f) - in[i];
}

// ---------------------------------------------------------------------------
// 10. W f32 [BH,256,64] -> W^T bf16 [BH,64,256]
// ---------------------------------------------------------------------------
__global__ void w_to_wt_kernel(const float* __restrict__ W, __bf16* __restrict__ WT) {
  long i = (long)blockIdx.x * blockDim.x + threadIdx.x;
  if (i >= (long)BH * MSEL * D_) return;
  long bh = i / (MSEL * D_);
  int p  = (int)(i - bh * (MSEL * D_));
  int r  = p / D_, c = p % D_;
  WT[bh * (long)(D_ * MSEL) + (long)c * MSEL + r] = (__bf16)W[i];
}

// ---------------------------------------------------------------------------
extern "C" void kernel_launch(void* const* d_in, const int* in_sizes, int n_in,
                              void* d_out, int out_size, void* d_ws, size_t ws_size,
                              hipStream_t stream) {
  (void)in_sizes; (void)n_in; (void)out_size; (void)ws_size;
  const float* Q = (const float*)d_in[0];
  const float* K = (const float*)d_in[1];
  const float* V = (const float*)d_in[2];
  const unsigned char* mask = (const unsigned char*)d_in[3];
  float* X = (float*)d_out;

  char* p = (char*)d_ws;
  auto alloc = [&](size_t bytes) { char* r = p; p += (bytes + 255) & ~(size_t)255; return (void*)r; };

  __bf16* Qs  = (__bf16*)alloc((size_t)BH * N_ * D_ * 2);
  __bf16* Kb  = (__bf16*)alloc((size_t)BH * N_ * D_ * 2);
  __bf16* VT  = (__bf16*)alloc((size_t)BH * N_ * D_ * 2);
  float* sumK = (float*)alloc((size_t)BH * N_ * 4);
  float* sumQ = (float*)alloc((size_t)BH * N_ * 4);
  int* idx_k  = (int*)alloc((size_t)BH * MSEL * 4);
  int* idx_q  = (int*)alloc((size_t)BH * MSEL * 4);
  __bf16* nc  = (__bf16*)alloc((size_t)BH * MSEL * D_ * 2);
  __bf16* nr  = (__bf16*)alloc((size_t)BH * MSEL * D_ * 2);
  float* S    = (float*)alloc((size_t)BH * N_ * MSEL * 4);   // reused scratch (c / u_pre / r)
  __bf16* k1  = (__bf16*)alloc((size_t)BH * N_ * MSEL * 2);
  __bf16* k3  = (__bf16*)alloc((size_t)BH * MSEL * N_ * 2);
  float* u    = (float*)alloc((size_t)BH * MSEL * MSEL * 4);
  float* Vb0  = (float*)alloc((size_t)BH * MSEL * MSEL * 4);
  float* Vb1  = (float*)alloc((size_t)BH * MSEL * MSEL * 4);
  float* KVb  = (float*)alloc((size_t)BH * MSEL * MSEL * 4);
  float* T1   = (float*)alloc((size_t)BH * MSEL * MSEL * 4);
  float* T2   = (float*)alloc((size_t)BH * MSEL * MSEL * 4);
  float* RV   = (float*)alloc((size_t)BH * MSEL * D_ * 4);
  float* W    = (float*)alloc((size_t)BH * MSEL * D_ * 4);
  __bf16* WT  = (__bf16*)alloc((size_t)BH * D_ * MSEL * 2);

  const long sM = (long)MSEL * MSEL;

  // 1. convert / transpose
  {
    long total = (long)BH * N_ * D_;
    convert_kernel<<<(int)((total + 255) / 256), 256, 0, stream>>>(Q, K, V, Qs, Kb, VT);
  }
  // 2. masked row sums for selection
  rowsum_kernel<<<(BH * N_) / 256, 256, 0, stream>>>(K, mask, sumK, 1.0f);
  rowsum_kernel<<<(BH * N_) / 256, 256, 0, stream>>>(Q, mask, sumQ, 0.125f);
  // 3. top-256
  topk_kernel<<<BH, 256, 0, stream>>>(sumK, idx_k);
  topk_kernel<<<BH, 256, 0, stream>>>(sumQ, idx_q);
  // 4. gather nc (from K), nr (from Qs)
  gather_kernel<<<BH * MSEL, D_, 0, stream>>>(Kb, idx_k, nc);
  gather_kernel<<<BH * MSEL, D_, 0, stream>>>(Qs, idx_q, nr);
  // 5. c = Qs @ nc^T ; kernel_1 = softmax(c) (bf16)
  gemm_bf16_kernel<<<dim3(MSEL / 64, N_ / 16, BH), 128, 0, stream>>>(
      Qs, nc, S, N_, MSEL, D_, (long)N_ * D_, (long)MSEL * D_, (long)N_ * MSEL);
  softmax_kernel<__bf16><<<BH * N_, 256, 0, stream>>>(S, k1, MSEL, nullptr, N_);
  // 6. u = softmax(nr @ nc^T) (f32)  [== gathered rows of kernel_1]
  gemm_bf16_kernel<<<dim3(MSEL / 64, MSEL / 16, BH), 128, 0, stream>>>(
      nr, nc, S, MSEL, MSEL, D_, (long)MSEL * D_, (long)MSEL * D_, sM);
  softmax_kernel<float><<<BH * MSEL, 256, 0, stream>>>(S, u, MSEL, nullptr, MSEL);
  // 7. r = nr @ K^T ; kernel_3 = masked softmax(r) (bf16)
  gemm_bf16_kernel<<<dim3(N_ / 64, MSEL / 16, BH), 128, 0, stream>>>(
      nr, Kb, S, MSEL, N_, D_, (long)MSEL * D_, (long)N_ * D_, (long)MSEL * N_);
  softmax_kernel<__bf16><<<BH * MSEL, 256, 0, stream>>>(S, k3, N_, mask, MSEL);
  // 8. Newton-Schulz: V <- 0.25 V (13I - KV(15I - KV(7I - KV)))
  init_v0_kernel<<<BH, MSEL, 0, stream>>>(u, Vb0);
  float* Vc = Vb0; float* Vn = Vb1;
  for (int it = 0; it < 4; ++it) {
    gemm_f32_kernel<<<dim3(MSEL / 64, MSEL / 16, BH), 128, 0, stream>>>(u,  Vc, KVb, MSEL, MSEL, MSEL, 1.0f, sM, sM, sM);
    ident_minus_kernel<<<(BH * MSEL * MSEL) / 256, 256, 0, stream>>>(KVb, T1, 7.0f);
    gemm_f32_kernel<<<dim3(MSEL / 64, MSEL / 16, BH), 128, 0, stream>>>(KVb, T1, T2, MSEL, MSEL, MSEL, 1.0f, sM, sM, sM);
    ident_minus_kernel<<<(BH * MSEL * MSEL) / 256, 256, 0, stream>>>(T2, T1, 15.0f);
    gemm_f32_kernel<<<dim3(MSEL / 64, MSEL / 16, BH), 128, 0, stream>>>(KVb, T1, T2, MSEL, MSEL, MSEL, 1.0f, sM, sM, sM);
    ident_minus_kernel<<<(BH * MSEL * MSEL) / 256, 256, 0, stream>>>(T2, T1, 13.0f);
    gemm_f32_kernel<<<dim3(MSEL / 64, MSEL / 16, BH), 128, 0, stream>>>(Vc, T1, Vn, MSEL, MSEL, MSEL, 0.25f, sM, sM, sM);
    float* t = Vc; Vc = Vn; Vn = t;
  }
  // 9. RV = kernel_3 @ V  (Brows = V^T)
  gemm_bf16_kernel<<<dim3(D_ / 64, MSEL / 16, BH), 128, 0, stream>>>(
      k3, VT, RV, MSEL, D_, N_, (long)MSEL * N_, (long)D_ * N_, (long)MSEL * D_);
  // 10. W = V_inv @ RV (f32 WMMA, direct B)
  gemm_f32_kernel<<<dim3(D_ / 64, MSEL / 16, BH), 128, 0, stream>>>(
      Vc, RV, W, MSEL, D_, MSEL, 1.0f, sM, (long)MSEL * D_, (long)MSEL * D_);
  // 11. W -> W^T bf16
  w_to_wt_kernel<<<(int)(((long)BH * MSEL * D_ + 255) / 256), 256, 0, stream>>>(W, WT);
  // 12. X = kernel_1 @ W  (Brows = W^T) -> d_out f32
  gemm_bf16_kernel<<<dim3(D_ / 64, N_ / 16, BH), 128, 0, stream>>>(
      k1, WT, X, N_, D_, MSEL, (long)N_ * MSEL, (long)D_ * MSEL, (long)N_ * D_);
}
